// SelfAttention_13804024889600
// MI455X (gfx1250) — matrix operand order repaired and checked by reading, not compiled, not run on hardware
//
#include <hip/hip_runtime.h>
#include <hip/hip_bf16.h>
#include <math.h>

typedef __bf16 bf16;
typedef __attribute__((ext_vector_type(16))) __bf16 v16bf;
typedef __attribute__((ext_vector_type(8)))  float  v8f;
typedef __attribute__((ext_vector_type(4)))  int    v4i;

#define EMB 1024
#define SEQ 2048
#define BATCH 4
#define HEADS 16
#define HDIM 64
#define QK_SCALE 0.17677669529663687f   /* 1024^-0.25 */

#define AS_GLOBAL __attribute__((address_space(1)))
#define AS_LOCAL  __attribute__((address_space(3)))

#if defined(__has_builtin)
#if __has_builtin(__builtin_amdgcn_global_load_async_to_lds_b128) && \
    __has_builtin(__builtin_amdgcn_s_wait_asynccnt)
#define HAVE_ASYNC_LDS 1
#endif
#endif
#ifndef HAVE_ASYNC_LDS
#define HAVE_ASYNC_LDS 0
#endif

__device__ __forceinline__ v8f zero8() {
    v8f z = {0.f, 0.f, 0.f, 0.f, 0.f, 0.f, 0.f, 0.f};
    return z;
}

__device__ __forceinline__ v8f wmma_bf16(v16bf a, v16bf b, v8f c) {
    // D = A(16x32 bf16) x B(32x16 bf16) + C(16x16 f32)
    return __builtin_amdgcn_wmma_f32_16x16x32_bf16(
        /*neg_a=*/false, a, /*neg_b=*/false, b,
        /*c_mod=*/(short)0, c, /*reuse_a=*/false, /*reuse_b=*/false);
}

union Frag { v16bf v; float4 f[2]; };

// A-fragment: 16x32 (MxK), lane holds row (lane&15); K packing per ISA 7.12.2:
// lanes 0-15: e0..7 = K0..7, e8..15 = K16..23 ; lanes 16-31: +8.
__device__ __forceinline__ v16bf load_a(const bf16* __restrict__ base, int ld,
                                        int row, int half) {
    Frag a;
    const bf16* p = base + (size_t)row * ld + half * 8;
    a.f[0] = *(const float4*)p;        // K = half*8 .. +7
    a.f[1] = *(const float4*)(p + 16); // K = 16+half*8 .. +7
    return a.v;
}

// B-fragment: 32x16 (KxN), lane n & n+16 hold column n;
// lanes 0-15: e0..7 = K0..7, e8..15 = K16..23 ; lanes 16-31: +8.
__device__ __forceinline__ v16bf load_b(const bf16* __restrict__ base, int ld,
                                        int col, int half) {
    Frag b;
    const bf16* p = base + (size_t)col * ld + half * 8;
    b.f[0] = *(const float4*)p;
    b.f[1] = *(const float4*)(p + 16);
    return b.v;
}

// 16-byte global -> LDS copy; async (ASYNCcnt) when the toolchain exposes it.
__device__ __forceinline__ void copy_b128_to_lds(const bf16* g, bf16* l) {
#if HAVE_ASYNC_LDS
    __builtin_amdgcn_global_load_async_to_lds_b128(
        (AS_GLOBAL v4i*)g, (AS_LOCAL v4i*)l, 0, 0);
#else
    *(float4*)l = *(const float4*)g;
#endif
}

__device__ __forceinline__ void wait_lds_copies() {
#if HAVE_ASYNC_LDS
    __builtin_amdgcn_s_wait_asynccnt(0);
#endif
}

__device__ __forceinline__ float redmax16(float t) {
#pragma unroll
    for (int off = 1; off < 16; off <<= 1) t = fmaxf(t, __shfl_xor(t, off, 32));
    return t;
}
__device__ __forceinline__ float redsum16(float t) {
#pragma unroll
    for (int off = 1; off < 16; off <<= 1) t += __shfl_xor(t, off, 32);
    return t;
}

// ---------------------------------------------------------------- conversions
__global__ void cvt_f32_bf16(const float* __restrict__ src,
                             bf16* __restrict__ dst, int n) {
    int i = blockIdx.x * blockDim.x + threadIdx.x;
    int stride = gridDim.x * blockDim.x;
    for (; i < n; i += stride) dst[i] = (bf16)src[i];
}

// ------------------------------------------------------------- QKV projection
// C = X(8192x1024) @ W^T ; per-wave tile 32(M) x 64(N): 8 WMMA per K-step,
// A-frags reused across 4 N-tiles, B-frags across 2 M-tiles (1.5 loads/WMMA).
// mode 0: store bf16 [b,h,s,d] scaled (Q,K). mode 2: store bf16 [b,h,d,s] (V^T).
__global__ __launch_bounds__(128)
void proj_qkv_kernel(const bf16* __restrict__ X, const bf16* __restrict__ W,
                     bf16* __restrict__ dst, int mode, float scale) {
    const int lane = threadIdx.x & 31;
    const int wave = threadIdx.x >> 5;
    const int row  = lane & 15;
    const int half = lane >> 4;
    const int mbase = blockIdx.x * 32;
    const int nbase = blockIdx.y * 256 + wave * 64;

    const bf16* A0 = X + (size_t)mbase * EMB;
    const bf16* A1 = A0 + (size_t)16 * EMB;

    v8f acc[2][4];
#pragma unroll
    for (int mt = 0; mt < 2; ++mt)
#pragma unroll
        for (int nt = 0; nt < 4; ++nt) acc[mt][nt] = zero8();

    for (int kk = 0; kk < EMB; kk += 32) {
        v16bf a0 = load_a(A0 + kk, EMB, row, half);
        v16bf a1 = load_a(A1 + kk, EMB, row, half);
#pragma unroll
        for (int nt = 0; nt < 4; ++nt) {
            v16bf b = load_b(W + (size_t)(nbase + nt * 16) * EMB + kk,
                             EMB, row, half);
            acc[0][nt] = wmma_bf16(a0, b, acc[0][nt]);
            acc[1][nt] = wmma_bf16(a1, b, acc[1][nt]);
        }
    }

#pragma unroll
    for (int mt = 0; mt < 2; ++mt)
#pragma unroll
        for (int nt = 0; nt < 4; ++nt)
#pragma unroll
            for (int r = 0; r < 8; ++r) {
                int m = mbase + mt * 16 + r + 8 * half;   // global token row
                int bidx = m >> 11, s = m & (SEQ - 1);
                int n = nbase + nt * 16 + row;            // global feature
                int h = n >> 6, dd = n & (HDIM - 1);
                bf16 v = (bf16)(acc[mt][nt][r] * scale);
                if (mode == 2)
                    dst[(((size_t)bidx * HEADS + h) * HDIM + dd) * SEQ + s] = v;
                else
                    dst[(((size_t)bidx * HEADS + h) * SEQ + s) * HDIM + dd] = v;
            }
}

// ------------------------------------------------------------ flash attention
// Q,K: [b,h,s,64] bf16 (pre-scaled); Vt: [b,h,64,s] bf16; O: [b,s,h*64] bf16.
// 4 waves/block handle 4 adjacent 16-query tiles of one (b,h); the identical
// K/V stream is staged once per block into LDS via async global->LDS copies.
__global__ __launch_bounds__(128)
void attn_kernel(const bf16* __restrict__ Q, const bf16* __restrict__ K,
                 const bf16* __restrict__ Vt, bf16* __restrict__ O) {
    const int lane = threadIdx.x & 31;
    const int wave = threadIdx.x >> 5;
    const int row  = lane & 15;
    const int half = lane >> 4;
    const int bh    = blockIdx.x;                 // b*16 + h
    const int qtile = blockIdx.y * 4 + wave;
    const int qbase = qtile * 16;

    const bf16* Qb = Q  + (size_t)bh * SEQ * HDIM + (size_t)qbase * HDIM;
    const bf16* Kg = K  + (size_t)bh * SEQ * HDIM;
    const bf16* Vg = Vt + (size_t)bh * HDIM * SEQ;

    // Q tile 16x64 fully in registers: two A-fragments (d-chunks of 32)
    v16bf qa0 = load_a(Qb,      HDIM, row, half);
    v16bf qa1 = load_a(Qb + 32, HDIM, row, half);

    v8f acc0 = zero8(), acc1 = zero8(), acc2 = zero8(), acc3 = zero8();
    float mrow[8], lrow[8];
#pragma unroll
    for (int r = 0; r < 8; ++r) { mrow[r] = -__builtin_inff(); lrow[r] = 0.f; }

    __shared__ __align__(16) bf16 Kt[32 * HDIM];        // [key][d]  4KB
    __shared__ __align__(16) bf16 Vs[HDIM * 32];        // [feat][key] 4KB
    __shared__ __align__(16) bf16 pstage[4][16 * 32];   // per-wave P transpose
    bf16* myp = pstage[wave];

    const int t = threadIdx.x;
    const int qend = qbase + 15;
    const int kmax = (blockIdx.y * 4 + 3) * 16 + 15;    // block-max causal bound

    for (int kb = 0; kb <= kmax; kb += 32) {
        // ---- cooperative async stage of K(32x64) and V(64x32) tiles
#pragma unroll
        for (int i = 0; i < 2; ++i) {
            int c = t + i * 128;                        // 256 16B chunks each
            int kr = c >> 3, kc = (c & 7) * 8;          // K tile: row 0..31
            copy_b128_to_lds(Kg + (size_t)(kb + kr) * HDIM + kc,
                             Kt + kr * HDIM + kc);
            int vr = c >> 2, vc = (c & 3) * 8;          // V tile: row 0..63
            copy_b128_to_lds(Vg + (size_t)vr * SEQ + kb + vc,
                             Vs + vr * 32 + vc);
        }
        wait_lds_copies();
        __syncthreads();

        if (kb <= qend) {
            // ---- scores S = Q K^T (16 q x 32 keys), contraction over d=64
            v8f s0 = zero8(), s1 = zero8();
            s0 = wmma_bf16(qa0, load_b(Kt,            HDIM, row, half), s0);
            s0 = wmma_bf16(qa1, load_b(Kt + 32,       HDIM, row, half), s0);
            s1 = wmma_bf16(qa0, load_b(Kt + 16 * HDIM,      HDIM, row, half), s1);
            s1 = wmma_bf16(qa1, load_b(Kt + 16 * HDIM + 32, HDIM, row, half), s1);

            // ---- causal mask + online softmax (per C-layout row)
#pragma unroll
            for (int r = 0; r < 8; ++r) {
                int qrow = qbase + r + 8 * half;
                int c0 = kb + row, c1 = kb + 16 + row;
                float v0 = (c0 <= qrow) ? s0[r] : -__builtin_inff();
                float v1 = (c1 <= qrow) ? s1[r] : -__builtin_inff();
                float mnew = fmaxf(mrow[r], redmax16(fmaxf(v0, v1)));
                float corr = __expf(mrow[r] - mnew);
                float p0 = __expf(v0 - mnew);
                float p1 = __expf(v1 - mnew);
                float sum = redsum16(p0 + p1);
                lrow[r] = lrow[r] * corr + sum;
                mrow[r] = mnew;
                acc0[r] *= corr; acc1[r] *= corr;
                acc2[r] *= corr; acc3[r] *= corr;
                int pr = r + 8 * half;                  // P: C-layout -> LDS
                myp[pr * 32 + row]      = (bf16)p0;
                myp[pr * 32 + 16 + row] = (bf16)p1;
            }

            // ---- O += P(16x32) @ V(32x64): reload P in A-layout from LDS
            Frag pa;
            const bf16* pp = myp + row * 32 + half * 8;
            pa.f[0] = *(const float4*)pp;
            pa.f[1] = *(const float4*)(pp + 16);

            acc0 = wmma_bf16(pa.v, load_b(Vs,           32, row, half), acc0);
            acc1 = wmma_bf16(pa.v, load_b(Vs + 16 * 32, 32, row, half), acc1);
            acc2 = wmma_bf16(pa.v, load_b(Vs + 32 * 32, 32, row, half), acc2);
            acc3 = wmma_bf16(pa.v, load_b(Vs + 48 * 32, 32, row, half), acc3);
        }
        __syncthreads();   // protect K/V tiles before next iteration's stage
    }

    // ---- finalize: O /= l ; store bf16 [b, s, h*64 + c*16 + col]
    const int b = bh >> 4, h = bh & 15;
#pragma unroll
    for (int r = 0; r < 8; ++r) {
        float inv = 1.0f / lrow[r];
        int s = qbase + r + 8 * half;
        size_t base = ((size_t)b * SEQ + s) * EMB + (size_t)h * HDIM + row;
        O[base +  0] = (bf16)(acc0[r] * inv);
        O[base + 16] = (bf16)(acc1[r] * inv);
        O[base + 32] = (bf16)(acc2[r] * inv);
        O[base + 48] = (bf16)(acc3[r] * inv);
    }
}

// --------------------------------------------------------- output projection
// out(f32) = Obf(8192x1024) @ Wu^T + bu ; same 32x64-per-wave tiling.
__global__ __launch_bounds__(128)
void out_proj_kernel(const bf16* __restrict__ Obf, const bf16* __restrict__ W,
                     const float* __restrict__ bias, float* __restrict__ out) {
    const int lane = threadIdx.x & 31;
    const int wave = threadIdx.x >> 5;
    const int row  = lane & 15;
    const int half = lane >> 4;
    const int mbase = blockIdx.x * 32;
    const int nbase = blockIdx.y * 256 + wave * 64;

    const bf16* A0 = Obf + (size_t)mbase * EMB;
    const bf16* A1 = A0 + (size_t)16 * EMB;

    v8f acc[2][4];
#pragma unroll
    for (int mt = 0; mt < 2; ++mt)
#pragma unroll
        for (int nt = 0; nt < 4; ++nt) acc[mt][nt] = zero8();

    for (int kk = 0; kk < EMB; kk += 32) {
        v16bf a0 = load_a(A0 + kk, EMB, row, half);
        v16bf a1 = load_a(A1 + kk, EMB, row, half);
#pragma unroll
        for (int nt = 0; nt < 4; ++nt) {
            v16bf b = load_b(W + (size_t)(nbase + nt * 16) * EMB + kk,
                             EMB, row, half);
            acc[0][nt] = wmma_bf16(a0, b, acc[0][nt]);
            acc[1][nt] = wmma_bf16(a1, b, acc[1][nt]);
        }
    }

#pragma unroll
    for (int nt = 0; nt < 4; ++nt) {
        int n = nbase + nt * 16 + row;
        float bv = bias[n];
#pragma unroll
        for (int mt = 0; mt < 2; ++mt)
#pragma unroll
            for (int r = 0; r < 8; ++r) {
                int m = mbase + mt * 16 + r + 8 * half;
                out[(size_t)m * EMB + n] = acc[mt][nt][r] + bv;
            }
    }
}

// ---------------------------------------------------------------------- host
extern "C" void kernel_launch(void* const* d_in, const int* in_sizes, int n_in,
                              void* d_out, int out_size, void* d_ws, size_t ws_size,
                              hipStream_t stream) {
    // setup_inputs() order: x, Wk, Wq, Wv, Wu, bu  (note: K before Q)
    const float* x  = (const float*)d_in[0];
    const float* Wk = (const float*)d_in[1];
    const float* Wq = (const float*)d_in[2];
    const float* Wv = (const float*)d_in[3];
    const float* Wu = (const float*)d_in[4];
    const float* bu = (const float*)d_in[5];
    float* out = (float*)d_out;

    const size_t NX = (size_t)BATCH * SEQ * EMB;   // 8388608
    const size_t NW = (size_t)EMB * EMB;           // 1048576

    char* ws = (char*)d_ws;
    size_t off = 0;
    bf16* xb  = (bf16*)(ws + off); off += NX * sizeof(bf16);
    bf16* wkb = (bf16*)(ws + off); off += NW * sizeof(bf16);
    bf16* wqb = (bf16*)(ws + off); off += NW * sizeof(bf16);
    bf16* wvb = (bf16*)(ws + off); off += NW * sizeof(bf16);
    bf16* wub = (bf16*)(ws + off); off += NW * sizeof(bf16);
    bf16* Qb  = (bf16*)(ws + off); off += NX * sizeof(bf16);
    bf16* Kb  = (bf16*)(ws + off); off += NX * sizeof(bf16);
    bf16* Vtb = (bf16*)(ws + off); off += NX * sizeof(bf16);
    bf16* Ob  = (bf16*)(ws + off); off += NX * sizeof(bf16);

    // 1) downconvert activations + weights to bf16
    cvt_f32_bf16<<<dim3(2048), dim3(256), 0, stream>>>(x,  xb,  (int)NX);
    cvt_f32_bf16<<<dim3(512),  dim3(256), 0, stream>>>(Wk, wkb, (int)NW);
    cvt_f32_bf16<<<dim3(512),  dim3(256), 0, stream>>>(Wq, wqb, (int)NW);
    cvt_f32_bf16<<<dim3(512),  dim3(256), 0, stream>>>(Wv, wvb, (int)NW);
    cvt_f32_bf16<<<dim3(512),  dim3(256), 0, stream>>>(Wu, wub, (int)NW);

    // 2) QKV projections (M=8192, N=1024, K=1024), 32x64 tile per wave
    dim3 gp(256, 4), bp(128);
    proj_qkv_kernel<<<gp, bp, 0, stream>>>(xb, wqb, Qb,  0, QK_SCALE);
    proj_qkv_kernel<<<gp, bp, 0, stream>>>(xb, wkb, Kb,  0, QK_SCALE);
    proj_qkv_kernel<<<gp, bp, 0, stream>>>(xb, wvb, Vtb, 2, 1.0f);

    // 3) causal flash attention: grid = (b*h, qtiles/4), 4 waves per block
    attn_kernel<<<dim3(BATCH * HEADS, SEQ / 64), dim3(128), 0, stream>>>(Qb, Kb, Vtb, Ob);

    // 4) output projection + bias -> fp32
    out_proj_kernel<<<gp, bp, 0, stream>>>(Ob, wub, bu, out);
}
